// Net_59536836657849
// MI455X (gfx1250) — matrix operand order repaired and checked by reading, not compiled, not run on hardware
//
#include <hip/hip_runtime.h>
#include <hip/hip_bf16.h>

// ---------------- output layout (floats), in reference return order ----------
#define NB 2048
static const size_t OFF_LOGP   = 0;                         // [2048,10]
static const size_t OFF_C1IN   = 20480;                     // [2048,1,28,28]
static const size_t OFF_C1OUT  = 1626112;                   // [2048,10,24,24]
static const size_t OFF_C2IN   = 13422592;                  // [2048,10,12,12]
static const size_t OFF_C2OUT  = 16371712;                  // [2048,20,8,8]
static const size_t OFF_FC1IN  = 18993152;                  // [2048,320]
static const size_t OFF_FC1OUT = 19648512;                  // [2048,50]
static const size_t OFF_FC2IN  = 19750912;                  // [2048,50]
static const size_t OFF_FC2OUT = 19853312;                  // [2048,10]

typedef __attribute__((ext_vector_type(16))) _Float16 v16h;
typedef __attribute__((ext_vector_type(8)))  _Float16 v8h;
typedef __attribute__((ext_vector_type(8)))  float    v8f;

__device__ __forceinline__ float quant8(float v) {
    // round to 2^-8 grid, RNE (matches jnp.round); *2^-8 is exact
    return rintf(v * 256.f) * (1.f / 256.f);
}
__device__ __forceinline__ float clamp87(float v) {
    return fminf(fmaxf(v, -8.f), 7.f);
}

// ============================================================================
// Kernel 1: fused quant -> conv1 -> pool -> conv2 -> pool. One WG per image;
// every intermediate lives in LDS; per-tap quant+clamp done in scaled-int
// floats (exact: |acc| <= 250*2048 < 2^24).
// ============================================================================
__global__ __launch_bounds__(256) void conv_pipeline(
    const float* __restrict__ x,
    const float* __restrict__ w1, const float* __restrict__ b1,
    const float* __restrict__ w2, const float* __restrict__ b2,
    float* __restrict__ out)
{
    __shared__ __align__(16) float s_x[784];
    __shared__ __align__(16) float s_c1[5760];
    __shared__ __align__(16) float s_p1[1440];
    __shared__ __align__(16) float s_c2[1280];
    __shared__ __align__(16) float s_w1[250];   // rint(w*256)  (scaled ints)
    __shared__ __align__(16) float s_w2[5000];  // rint(w*256)
    __shared__ float s_b1[10];                  // quant(b,8)
    __shared__ float s_b2[20];

    const int b   = blockIdx.x;
    const int tid = threadIdx.x;
    const float IS = 1.f / 256.f;

    // stage + quantize input; write conv1_input
    for (int i = tid; i < 784; i += 256) {
        float q = quant8(x[(size_t)b * 784 + i]);
        s_x[i] = q;
        out[OFF_C1IN + (size_t)b * 784 + i] = q;
    }
    for (int i = tid; i < 250;  i += 256) s_w1[i] = rintf(w1[i] * 256.f);
    for (int i = tid; i < 5000; i += 256) s_w2[i] = rintf(w2[i] * 256.f);
    if (tid < 10) s_b1[tid] = quant8(b1[tid]);
    if (tid < 20) s_b2[tid] = quant8(b2[tid]);
    __syncthreads();

    // ---- conv1: 10 x 24 x 24 ----
    for (int idx = tid; idx < 5760; idx += 256) {
        int oc = idx / 576, pos = idx % 576, oy = pos / 24, ox = pos % 24;
        const float* wb = s_w1 + oc * 25;
        const float* xb = s_x + oy * 28 + ox;
        float acc = 0.f;
        #pragma unroll
        for (int p = 0; p < 5; ++p)
            #pragma unroll
            for (int q = 0; q < 5; ++q) {
                float t = rintf(xb[p * 28 + q] * wb[p * 5 + q]);  // x*(w*256)
                t = fminf(fmaxf(t, -2048.f), 1792.f);             // clamp*256
                acc += t;
            }
        float o = acc * IS + s_b1[oc];     // exact grid value; quant idempotent
        s_c1[idx] = o;
        out[OFF_C1OUT + (size_t)b * 5760 + idx] = o;
    }
    __syncthreads();

    // ---- pool1 + relu -> conv2_input: 10 x 12 x 12 ----
    for (int idx = tid; idx < 1440; idx += 256) {
        int oc = idx / 144, pos = idx % 144, oy = pos / 12, ox = pos % 12;
        const float* p = s_c1 + oc * 576 + oy * 48 + ox * 2;
        float m = fmaxf(fmaxf(p[0], p[1]), fmaxf(p[24], p[25]));
        m = fmaxf(m, 0.f);
        s_p1[idx] = m;
        out[OFF_C2IN + (size_t)b * 1440 + idx] = m;
    }
    __syncthreads();

    // ---- conv2: 20 x 8 x 8 ----
    for (int idx = tid; idx < 1280; idx += 256) {
        int oc = idx / 64, pos = idx % 64, oy = pos / 8, ox = pos % 8;
        float acc = 0.f;
        for (int ic = 0; ic < 10; ++ic) {
            const float* xb = s_p1 + ic * 144 + oy * 12 + ox;
            const float* wb = s_w2 + oc * 250 + ic * 25;
            #pragma unroll
            for (int p = 0; p < 5; ++p)
                #pragma unroll
                for (int q = 0; q < 5; ++q) {
                    float t = rintf(xb[p * 12 + q] * wb[p * 5 + q]);
                    t = fminf(fmaxf(t, -2048.f), 1792.f);
                    acc += t;
                }
        }
        float o = acc * IS + s_b2[oc];
        s_c2[idx] = o;
        out[OFF_C2OUT + (size_t)b * 1280 + idx] = o;
    }
    __syncthreads();

    // ---- pool2 + relu -> fc1_input: 320 ----
    for (int idx = tid; idx < 320; idx += 256) {
        int oc = idx / 16, pos = idx % 16, oy = pos / 4, ox = pos % 4;
        const float* p = s_c2 + oc * 64 + oy * 16 + ox * 2;
        float m = fmaxf(fmaxf(p[0], p[1]), fmaxf(p[8], p[9]));
        m = fmaxf(m, 0.f);
        out[OFF_FC1IN + (size_t)b * 320 + idx] = m;
    }
}

// ============================================================================
// Kernel 2 (prep): quantize+clamp FC weights/biases into padded f16 matrices.
// wq1: [64][320] f16 (rows 50..63 zero)   wq2: [16][64] f16 (n>=10 / k>=50 zero)
// ============================================================================
__global__ __launch_bounds__(256) void prep_weights(
    const float* __restrict__ fc1w, const float* __restrict__ fc1b,
    const float* __restrict__ fc2w, const float* __restrict__ fc2b,
    _Float16* __restrict__ wq1, _Float16* __restrict__ wq2,
    float* __restrict__ bq1, float* __restrict__ bq2)
{
    int tid = threadIdx.x;
    for (int i = tid; i < 64 * 320; i += 256) {
        int n = i / 320, k = i % 320;
        float v = 0.f;
        if (n < 50) v = clamp87(quant8(fc1w[n * 320 + k]));
        wq1[i] = (_Float16)v;
    }
    for (int i = tid; i < 16 * 64; i += 256) {
        int n = i / 64, k = i % 64;
        float v = 0.f;
        if (n < 10 && k < 50) v = clamp87(quant8(fc2w[n * 50 + k]));
        wq2[i] = (_Float16)v;
    }
    if (tid < 64) bq1[tid] = (tid < 50) ? clamp87(quant8(fc1b[tid])) : 0.f;
    if (tid < 16) bq2[tid] = (tid < 10) ? clamp87(quant8(fc2b[tid])) : 0.f;
}

// ============================================================================
// Kernel 3: fc1 (WMMA, M=16 batch rows per WG, 4 waves x 16-col N tiles,
// K=320 -> 10x v_wmma_f32_16x16x32_f16) -> clamp/quant/relu -> fc2 (2 WMMAs
// on wave 0) -> log_softmax. All FC operands are exact in f16.
// ============================================================================
__global__ __launch_bounds__(128) void fc_pipeline(
    const _Float16* __restrict__ wq1, const _Float16* __restrict__ wq2,
    const float* __restrict__ bq1, const float* __restrict__ bq2,
    float* __restrict__ out)
{
    __shared__ __align__(32) _Float16 s_xh[16 * 320];  // fc1 input tile, f16
    __shared__ __align__(32) _Float16 s_x2[16 * 64];   // fc2 input tile, f16
    __shared__ __align__(16) float    s_lg[16 * 16];   // fc2 logits

    const int tid  = threadIdx.x;
    const int wave = tid >> 5;
    const int lane = tid & 31;
    const int row0 = blockIdx.x * 16;

    __builtin_prefetch(wq1 + (size_t)(wave * 16) * 320, 0, 0);

    // stage 16x320 fc1_input, clamp to [-8,7] (round_max), convert f16 (exact)
    const float* gx = out + OFF_FC1IN + (size_t)row0 * 320;
    for (int i = tid; i < 16 * 320; i += 128) {
        s_xh[i] = (_Float16)clamp87(gx[i]);
    }
    __syncthreads();

    const int m  = lane & 15;      // A row within tile / B column within tile
    const int g  = lane >> 4;      // lane half-group
    const int n  = wave * 16 + m;  // global (padded) output column 0..63

    // ---- fc1: K=320, 10 WMMA steps ----
    v8f acc = {};
    #pragma unroll
    for (int kc = 0; kc < 320; kc += 32) {
        v16h a = *(const v16h*)(s_xh + m * 320 + kc + g * 16);
        v16h bf = *(const v16h*)(wq1 + (size_t)n * 320 + kc + g * 16);
        acc = __builtin_amdgcn_wmma_f32_16x16x32_f16(
                  false, a, false, bf, (short)0, acc, false, false);
    }

    // epilogue: +bias, clamp, quant -> fc1_output; relu -> fc2_input
    {
        float bias = bq1[n];
        #pragma unroll
        for (int r = 0; r < 8; ++r) {
            int mr = r + 8 * g;                        // D row: lanes16-31 => +8
            float v = acc[r] + bias;
            v = quant8(clamp87(v));
            float rv = fmaxf(v, 0.f);
            if (n < 50) {
                out[OFF_FC1OUT + (size_t)(row0 + mr) * 50 + n] = v;
                out[OFF_FC2IN  + (size_t)(row0 + mr) * 50 + n] = rv;
            }
            s_x2[mr * 64 + n] = (n < 50) ? (_Float16)rv : (_Float16)0.f;
        }
    }
    __syncthreads();

    // ---- fc2: K=64 (50 padded), N=16 (10 padded): wave 0 only ----
    if (wave == 0) {
        v8f acc2 = {};
        #pragma unroll
        for (int kc = 0; kc < 64; kc += 32) {
            v16h a = *(const v16h*)(s_x2 + m * 64 + kc + g * 16);
            v16h bf = *(const v16h*)(wq2 + (size_t)m * 64 + kc + g * 16);
            acc2 = __builtin_amdgcn_wmma_f32_16x16x32_f16(
                       false, a, false, bf, (short)0, acc2, false, false);
        }
        float bias2 = bq2[m];
        #pragma unroll
        for (int r = 0; r < 8; ++r) {
            int mr = r + 8 * g;
            float v = acc2[r] + bias2;
            v = quant8(clamp87(v));
            s_lg[mr * 16 + m] = v;
            if (m < 10) out[OFF_FC2OUT + (size_t)(row0 + mr) * 10 + m] = v;
        }
    }
    __syncthreads();

    // ---- log_softmax over 10 classes, one row per lane ----
    if (tid < 16) {
        float vals[10];
        float mx = -3.402823e38f;
        #pragma unroll
        for (int j = 0; j < 10; ++j) {
            vals[j] = s_lg[tid * 16 + j];
            mx = fmaxf(mx, vals[j]);
        }
        float s = 0.f;
        #pragma unroll
        for (int j = 0; j < 10; ++j) s += expf(vals[j] - mx);
        float l = logf(s);
        #pragma unroll
        for (int j = 0; j < 10; ++j)
            out[OFF_LOGP + (size_t)(row0 + tid) * 10 + j] = vals[j] - mx - l;
    }
}

// ============================================================================
extern "C" void kernel_launch(void* const* d_in, const int* in_sizes, int n_in,
                              void* d_out, int out_size, void* d_ws, size_t ws_size,
                              hipStream_t stream) {
    const float* x   = (const float*)d_in[0];
    const float* w1  = (const float*)d_in[1];
    const float* b1  = (const float*)d_in[2];
    const float* w2  = (const float*)d_in[3];
    const float* b2  = (const float*)d_in[4];
    const float* f1w = (const float*)d_in[5];
    const float* f1b = (const float*)d_in[6];
    const float* f2w = (const float*)d_in[7];
    const float* f2b = (const float*)d_in[8];
    float* out = (float*)d_out;

    // workspace carving: f16 weight matrices + f32 biases (~43 KB)
    char* ws = (char*)d_ws;
    _Float16* wq1 = (_Float16*)(ws);                 // 64*320*2 = 40960 B
    _Float16* wq2 = (_Float16*)(ws + 40960);         // 16*64*2  =  2048 B
    float*    bq1 = (float*)(ws + 43008);            // 64*4
    float*    bq2 = (float*)(ws + 43264);            // 16*4

    prep_weights<<<1, 256, 0, stream>>>(f1w, f1b, f2w, f2b, wq1, wq2, bq1, bq2);
    conv_pipeline<<<NB, 256, 0, stream>>>(x, w1, b1, w2, b2, out);
    fc_pipeline<<<NB / 16, 128, 0, stream>>>(wq1, wq2, bq1, bq2, out);
}
